// DrugGraphEmbedding_42047729828323
// MI455X (gfx1250) — matrix operand order repaired and checked by reading, not compiled, not run on hardware
//
#include <hip/hip_runtime.h>
#include <cstdint>

typedef __attribute__((ext_vector_type(16))) _Float16 v16h;
typedef __attribute__((ext_vector_type(8)))  _Float16 v8h;
typedef __attribute__((ext_vector_type(8)))  float    v8f;

#define NN 49152      // B*S nodes
#define DD 256        // input dim
#define FF 512        // feature dim (2*D = GDIM)
#define KK 1024       // GEMM K (concat of mean|self)
#define EE 196608     // directed edges
#define BB 1024       // graphs
#define SS 48         // nodes per graph

#define LDS_STRIDE 136            // 128 K-halves + 8 pad halves (272B row: conflict-free b128)
#define LDS_BUF    (128 * LDS_STRIDE)

// ---------------- graph preprocessing kernels (f32, bandwidth-bound) ----------------

__global__ void k_count(const long long* __restrict__ ei,
                        float* __restrict__ deg, float* __restrict__ cnt) {
  int e = blockIdx.x * blockDim.x + threadIdx.x;
  if (e >= EE) return;
  int r = (int)ei[e];
  int c = (int)ei[EE + e];
  atomicAdd(deg + r, 1.0f);
  atomicAdd(cnt + c, 1.0f);
}

__global__ void k_dinv(const float* __restrict__ deg, float* __restrict__ dinv) {
  int i = blockIdx.x * blockDim.x + threadIdx.x;
  if (i >= NN) return;
  float d = deg[i];
  dinv[i] = (d > 0.0f) ? (1.0f / sqrtf(d)) : 0.0f;
}

// lap[row] += -(dinv[row]*dinv[col]) * x[col]   (x: [N,256])
__global__ void k_lap(const long long* __restrict__ ei, const float* __restrict__ x,
                      const float* __restrict__ dinv, float* __restrict__ lap) {
  int t = blockIdx.x * blockDim.x + threadIdx.x;   // EE * 64 threads
  int e  = t >> 6;
  int f4 = (t & 63) * 4;
  if (e >= EE) return;
  int r = (int)ei[e];
  int c = (int)ei[EE + e];
  float w = -(dinv[r] * dinv[c]);
  const float4 xv = *(const float4*)(x + (size_t)c * DD + f4);
  float* dst = lap + (size_t)r * DD + f4;
  atomicAdd(dst + 0, w * xv.x);
  atomicAdd(dst + 1, w * xv.y);
  atomicAdd(dst + 2, w * xv.z);
  atomicAdd(dst + 3, w * xv.w);
}

// xcomb[n, 0:256] = x ; xcomb[n, 256:512] = x + lap
__global__ void k_xcomb(const float* __restrict__ x, const float* __restrict__ lap,
                        float* __restrict__ xcomb) {
  int t = blockIdx.x * blockDim.x + threadIdx.x;   // NN * 64 threads
  int n  = t >> 6;
  int f4 = (t & 63) * 4;
  if (n >= NN) return;
  const float4 xv = *(const float4*)(x   + (size_t)n * DD + f4);
  const float4 lv = *(const float4*)(lap + (size_t)n * DD + f4);
  float4 sv; sv.x = xv.x + lv.x; sv.y = xv.y + lv.y; sv.z = xv.z + lv.z; sv.w = xv.w + lv.w;
  *(float4*)(xcomb + (size_t)n * FF + f4)      = xv;
  *(float4*)(xcomb + (size_t)n * FF + DD + f4) = sv;
}

// agg[col] += src[row]   (src: [N,512])
__global__ void k_scatter512(const long long* __restrict__ ei, const float* __restrict__ src,
                             float* __restrict__ agg) {
  int t = blockIdx.x * blockDim.x + threadIdx.x;   // EE * 128 threads
  int e  = t >> 7;
  int f4 = (t & 127) * 4;
  if (e >= EE) return;
  int r = (int)ei[e];
  int c = (int)ei[EE + e];
  const float4 sv = *(const float4*)(src + (size_t)r * FF + f4);
  float* dst = agg + (size_t)c * FF + f4;
  atomicAdd(dst + 0, sv.x);
  atomicAdd(dst + 1, sv.y);
  atomicAdd(dst + 2, sv.z);
  atomicAdd(dst + 3, sv.w);
}

// A16[n, 0:512] = f16(agg/max(cnt,1)) ; A16[n, 512:1024] = f16(src)
__global__ void k_abuild(const float* __restrict__ agg, const float* __restrict__ cnt,
                         const float* __restrict__ src, _Float16* __restrict__ A16) {
  int t = blockIdx.x * blockDim.x + threadIdx.x;   // NN * 512 threads
  if (t >= NN * FF) return;
  int n = t >> 9;
  int f = t & 511;
  float inv = 1.0f / fmaxf(cnt[n], 1.0f);
  A16[(size_t)n * KK + f]      = (_Float16)(agg[t] * inv);
  A16[(size_t)n * KK + FF + f] = (_Float16)(src[t]);
}

// Wcat[g, 0:512] = Wl[g,:], Wcat[g, 512:1024] = Wr[g,:] as f16
__global__ void k_wcat(const float* __restrict__ Wl, const float* __restrict__ Wr,
                       _Float16* __restrict__ Wcat) {
  int t = blockIdx.x * blockDim.x + threadIdx.x;   // 512*1024 threads
  if (t >= FF * KK) return;
  int g = t >> 10;
  int k = t & 1023;
  float v = (k < FF) ? Wl[(size_t)g * FF + k] : Wr[(size_t)g * FF + (k - FF)];
  Wcat[t] = (_Float16)v;
}

// ---------------- WMMA GEMM: out[N,512] = act(A[N,1024] @ Wcat[512,1024]^T + bias) ----
// block = 8 waves = 8 M-tiles (128 rows) x one 128-col N-group.
// W K-chunks (128 cols x 128 K-halves = 32KB) stream through LDS via async copy,
// double-buffered; each wave runs 4 K-steps x 8 WMMA per chunk from LDS.
__global__ __launch_bounds__(256) void k_gemm(const _Float16* __restrict__ A,
                                              const _Float16* __restrict__ W,
                                              const float* __restrict__ bias,
                                              float* __restrict__ out, int do_gelu) {
  __shared__ __attribute__((aligned(32))) _Float16 smem[2 * LDS_BUF];

  const int tid   = threadIdx.x;
  const int lane  = tid & 31;
  const int wv    = tid >> 5;                 // 0..7
  const int mb    = blockIdx.x >> 2;          // 0..383
  const int ngrp  = blockIdx.x & 3;
  const int mtile = mb * 8 + wv;              // 0..3071
  const int m     = lane & 15;
  const int hi    = (lane >> 4) & 1;
  const int colb  = ngrp * 128;
  const int kgoff = hi * 8;

  v8f acc[8] = {};

  // A-fragment base (per ISA layout: lanes0-15 K={0..7,16..23}, lanes16-31 K={8..15,24..31})
  const _Float16* arow = A + (size_t)(mtile * 16 + m) * KK + hi * 8;

  const unsigned lbase = (unsigned)(uintptr_t)(&smem[0]);

  // issue async copy of W chunk `c` into LDS buffer `b`
  auto issue_copy = [&](int c, int b) {
    const int kc = c * 128;
#pragma unroll
    for (int i = 0; i < 8; ++i) {
      int segid = tid + i * 256;              // 0..2047 : 128 rows x 16 x 16B segs
      int row = segid >> 4;
      int seg = segid & 15;
      unsigned laddr = lbase + (unsigned)((b * LDS_BUF + row * LDS_STRIDE + seg * 8) * 2);
      uint64_t gaddr = (uint64_t)(uintptr_t)(W + (size_t)(colb + row) * KK + kc + seg * 8);
      asm volatile("global_load_async_to_lds_b128 %0, %1, off"
                   :: "v"(laddr), "v"(gaddr) : "memory");
    }
  };

  issue_copy(0, 0);

  for (int c = 0; c < 8; ++c) {
    if (c < 7) {
      issue_copy(c + 1, (c + 1) & 1);
      asm volatile("s_wait_asynccnt 0x8" ::: "memory");   // chunk c complete (in-order)
    } else {
      asm volatile("s_wait_asynccnt 0x0" ::: "memory");
    }
    __syncthreads();

    const int kc = c * 128;
    // prefetch the 4 A-fragments of this chunk into registers (8 x b128)
    v8h areg[8];
#pragma unroll
    for (int s = 0; s < 4; ++s) {
      areg[2 * s]     = *(const v8h*)(arow + kc + s * 32);
      areg[2 * s + 1] = *(const v8h*)(arow + kc + s * 32 + 16);
    }

    const _Float16* sb = &smem[(c & 1) * LDS_BUF];
#pragma unroll
    for (int s = 0; s < 4; ++s) {
      v16h av = __builtin_shufflevector(areg[2 * s], areg[2 * s + 1],
                                        0,1,2,3,4,5,6,7,8,9,10,11,12,13,14,15);
#pragma unroll
      for (int nt = 0; nt < 8; ++nt) {
        const _Float16* bp = sb + (nt * 16 + m) * LDS_STRIDE + s * 32 + kgoff;
        v8h blo = *(const v8h*)bp;
        v8h bhi = *(const v8h*)(bp + 16);
        v16h bv = __builtin_shufflevector(blo, bhi,
                                          0,1,2,3,4,5,6,7,8,9,10,11,12,13,14,15);
        acc[nt] = __builtin_amdgcn_wmma_f32_16x16x32_f16(
            /*neg_a=*/false, av, /*neg_b=*/false, bv,
            /*c_mod=*/(short)0, acc[nt], /*reuse_a=*/false, /*reuse_b=*/false);
      }
    }
    __syncthreads();   // all waves done reading this buffer before it is refilled
  }

  // C layout: VGPR r -> row (r + 8*hi), col = colb + nt*16 + (lane%16)
  const int row0 = mtile * 16 + hi * 8;
#pragma unroll
  for (int nt = 0; nt < 8; ++nt) {
    const int col = colb + nt * 16 + m;
    const float bv = bias[col];
#pragma unroll
    for (int r = 0; r < 8; ++r) {
      float v = acc[nt][r] + bv;
      if (do_gelu) v = 0.5f * v * (1.0f + erff(v * 0.70710678118654752f));
      out[(size_t)(row0 + r) * FF + col] = v;
    }
  }
}

// out[b,g] = mean over 48 nodes
__global__ void k_pool(const float* __restrict__ G, float* __restrict__ out) {
  int t = blockIdx.x * blockDim.x + threadIdx.x;   // BB*512 threads
  if (t >= BB * FF) return;
  int b = t >> 9;
  int g = t & 511;
  float s = 0.0f;
  const float* p = G + (size_t)b * SS * FF + g;
#pragma unroll 8
  for (int i = 0; i < SS; ++i) s += p[(size_t)i * FF];
  out[t] = s * (1.0f / (float)SS);
}

// ---------------- host launcher ----------------

extern "C" void kernel_launch(void* const* d_in, const int* in_sizes, int n_in,
                              void* d_out, int out_size, void* d_ws, size_t ws_size,
                              hipStream_t stream) {
  (void)in_sizes; (void)n_in; (void)out_size; (void)ws_size;
  const float*     x   = (const float*)d_in[0];
  const long long* ei  = (const long long*)d_in[1];   // int64 edge_index [2,E]
  const float*     Wl1 = (const float*)d_in[2];
  const float*     Wr1 = (const float*)d_in[3];
  const float*     b1  = (const float*)d_in[4];
  const float*     Wl2 = (const float*)d_in[5];
  const float*     Wr2 = (const float*)d_in[6];
  const float*     b2  = (const float*)d_in[7];
  float*           out = (float*)d_out;

  char* ws = (char*)d_ws;
  size_t off = 0;
  auto carve = [&](size_t bytes) -> void* {
    void* p = ws + off;
    off = (off + bytes + 255) & ~(size_t)255;
    return p;
  };
  float*    deg   = (float*)carve((size_t)NN * 4);
  float*    cnt   = (float*)carve((size_t)NN * 4);
  float*    dinv  = (float*)carve((size_t)NN * 4);
  float*    agg   = (float*)carve((size_t)NN * FF * 4);   // also hosts lap (first N*256 floats)
  float*    xcomb = (float*)carve((size_t)NN * FF * 4);   // reused as GEMM2 output G
  _Float16* A16   = (_Float16*)carve((size_t)NN * KK * 2);
  float*    H32   = (float*)carve((size_t)NN * FF * 4);
  _Float16* Wc1   = (_Float16*)carve((size_t)FF * KK * 2);
  _Float16* Wc2   = (_Float16*)carve((size_t)FF * KK * 2);
  float* lap = agg;      // alias: lap used before agg is needed
  float* G   = xcomb;    // alias: xcomb dead after A1 build

  hipMemsetAsync(deg, 0, (size_t)NN * 4, stream);
  hipMemsetAsync(cnt, 0, (size_t)NN * 4, stream);
  hipMemsetAsync(lap, 0, (size_t)NN * DD * 4, stream);

  k_count<<<EE / 256, 256, 0, stream>>>(ei, deg, cnt);
  k_dinv<<<NN / 256, 256, 0, stream>>>(deg, dinv);
  k_lap<<<(EE * 64) / 256, 256, 0, stream>>>(ei, x, dinv, lap);
  k_xcomb<<<(NN * 64) / 256, 256, 0, stream>>>(x, lap, xcomb);
  k_wcat<<<(FF * KK) / 256, 256, 0, stream>>>(Wl1, Wr1, Wc1);
  k_wcat<<<(FF * KK) / 256, 256, 0, stream>>>(Wl2, Wr2, Wc2);

  // conv1: mean-aggregate x_comb, build A1, GEMM + GELU
  hipMemsetAsync(agg, 0, (size_t)NN * FF * 4, stream);
  k_scatter512<<<(EE * 128) / 256, 256, 0, stream>>>(ei, xcomb, agg);
  k_abuild<<<(NN * FF) / 256, 256, 0, stream>>>(agg, cnt, xcomb, A16);
  k_gemm<<<1536, 256, 0, stream>>>(A16, Wc1, b1, H32, 1);

  // conv2: mean-aggregate H, build A2, GEMM
  hipMemsetAsync(agg, 0, (size_t)NN * FF * 4, stream);
  k_scatter512<<<(EE * 128) / 256, 256, 0, stream>>>(ei, H32, agg);
  k_abuild<<<(NN * FF) / 256, 256, 0, stream>>>(agg, cnt, H32, A16);
  k_gemm<<<1536, 256, 0, stream>>>(A16, Wc2, b2, G, 0);

  // global mean pool
  k_pool<<<(BB * FF) / 256, 256, 0, stream>>>(G, out);
}
